// MultiModalFusionNetwork_88759794139604
// MI455X (gfx1250) — matrix operand order repaired and checked by reading, not compiled, not run on hardware
//
#include <hip/hip_runtime.h>
#include <hip/hip_bf16.h>
#include <cstdint>
#include <cstddef>

// ---------------- model dims ----------------
#define L_      6
#define D_      512
#define H_      8
#define HD_     64
#define FF_     2048
#define IN_     50
#define MAXREL_ 128
#define NREL_   257      // 2*MAXREL+1
#define NRELP_  272      // padded to multiple of 16
#define B_      16
#define S_      512
#define ROWS_   (B_*S_)  // 8192
#define EPS_    1e-5f

typedef __bf16 v16bf __attribute__((ext_vector_type(16)));
typedef float  v8f   __attribute__((ext_vector_type(8)));

union FragB { unsigned u[8]; v16bf v; };
union FragF { float f[8];    v8f   v; };
union U4x2  { uint4 q[2]; unsigned short s[16]; };

// gfx1250 async global->LDS copy (ASYNCcnt-tracked).  INST_OFFSET adds to BOTH
// the global and the LDS address (ISA 10.x / CDNA5 ISA), so one
// vdst/vaddr pair covers a contiguous multi-chunk transfer.
#define ASYNC_B128(ldsoff, gaddr, IOFF)                                         \
  asm volatile("global_load_async_to_lds_b128 %0, %1, off offset:" #IOFF        \
               :: "v"(ldsoff), "v"(gaddr) : "memory")
#define WAIT_ASYNC() asm volatile("s_wait_asynccnt 0x0" ::: "memory")
#define WAIT_DS()    asm volatile("s_wait_dscnt 0x0" ::: "memory")

__device__ __forceinline__ unsigned short f2bf(float x) {
  union { float f; unsigned u; } c; c.f = x;
  unsigned r = c.u + 0x7FFFu + ((c.u >> 16) & 1u);   // RNE
  return (unsigned short)(r >> 16);
}
__device__ __forceinline__ float bf2f(unsigned short h) {
  union { float f; unsigned u; } c; c.u = ((unsigned)h) << 16;
  return c.f;
}
// 16-bit A-matrix 16x32 K index for VGPR p, lane-half (ISA 7.12.2)
__device__ __forceinline__ int kpA(int p, int half) { return ((p & 3) * 2) + ((p >> 2) * 16) + half * 8; }
// 16-bit B-matrix 32x16 K index (same K order as A: lanes 0-15: K=0..7,16..23; lanes 16-31: K=8..15,24..31)
__device__ __forceinline__ int kpB(int p, int half) { return ((p & 3) * 2) + ((p >> 2) * 16) + half * 8; }
__device__ __forceinline__ unsigned ldsoff(const void* p) { return (unsigned)(uintptr_t)p; }

// ---------------- weight conversion ----------------
__global__ void k_transpose_bf16(const float* __restrict__ in, unsigned short* __restrict__ out,
                                 int K, int N) {
  int idx = blockIdx.x * 256 + threadIdx.x;
  if (idx >= K * N) return;
  int k = idx / N, n = idx % N;
  out[(size_t)n * K + k] = f2bf(in[idx]);
}
__global__ void k_copy_bf16(const float* __restrict__ in, unsigned short* __restrict__ out, int n) {
  int idx = blockIdx.x * 256 + threadIdx.x;
  if (idx < n) out[idx] = f2bf(in[idx]);
}

// ---------------- input projection + sinusoidal PE ----------------
__global__ __launch_bounds__(128) void k_input(const float* __restrict__ x, const float* __restrict__ w,
                                               const float* __restrict__ b, float* __restrict__ h,
                                               unsigned short* __restrict__ hb) {
  __shared__ float xs[IN_];
  int row = blockIdx.x, tid = threadIdx.x;
  if (tid < IN_) xs[tid] = x[(size_t)row * IN_ + tid];
  __syncthreads();
  int s = row & (S_ - 1);
  for (int j = 0; j < 4; ++j) {
    int c = tid + 128 * j;
    float acc = b[c];
    for (int k = 0; k < IN_; ++k) acc += xs[k] * w[(size_t)k * D_ + c];
    int ce = c & ~1;
    float div = __expf(-logf(10000.0f) * (float)ce / (float)D_);
    float ang = (float)s * div;
    acc += (c & 1) ? __cosf(ang) : __sinf(ang);
    h[(size_t)row * D_ + c] = acc;
    hb[(size_t)row * D_ + c] = f2bf(acc);
  }
}

// ---------------- generic bf16 WMMA GEMM: C = act(scale*A@B + bias) ----------------
// A [M,K] bf16 row-major, BT [N,K] bf16 row-major (i.e. B transposed)
// Async double-buffered tile pipeline: loads for step k+1 overlap WMMA of step k.
#define GBM 128
#define GBN 64
#define GBK 32
#define ALD 40
#define BLD 40
__global__ __launch_bounds__(128) void k_gemm(const unsigned short* __restrict__ A,
                                              const unsigned short* __restrict__ BT,
                                              const float* __restrict__ bias,
                                              float scale, int relu,
                                              int M, int N, int K,
                                              float* __restrict__ outF,
                                              unsigned short* __restrict__ outB) {
  __shared__ __align__(16) unsigned short as[2][GBM * ALD];
  __shared__ __align__(16) unsigned short bs[2][GBN * BLD];
  int tid = threadIdx.x;
  int wave = tid >> 5, lane = tid & 31;
  int half = lane >> 4, l16 = lane & 15;
  int bm0 = blockIdx.y * GBM, bn0 = blockIdx.x * GBN;

  const unsigned short* aptr = A + (size_t)(bm0 + tid) * K;          // this thread's A row
  int bn = tid >> 1, bch = (tid & 1) * 16;
  const unsigned short* bptr = BT + (size_t)(bn0 + bn) * K + bch;    // this thread's B chunk
  unsigned aoff[2] = { ldsoff(&as[0][tid * ALD]), ldsoff(&as[1][tid * ALD]) };
  unsigned boff[2] = { ldsoff(&bs[0][bn * BLD + bch]), ldsoff(&bs[1][bn * BLD + bch]) };

  FragF acc[2][4];
  for (int i = 0; i < 2; i++)
    for (int j = 0; j < 4; j++)
      for (int e = 0; e < 8; e++) acc[i][j].f[e] = 0.f;

  // prologue: async-load first tile into buffer 0
  {
    unsigned long long ga = (unsigned long long)(uintptr_t)aptr;
    ASYNC_B128(aoff[0], ga, 0);  ASYNC_B128(aoff[0], ga, 16);
    ASYNC_B128(aoff[0], ga, 32); ASYNC_B128(aoff[0], ga, 48);
    unsigned long long gb = (unsigned long long)(uintptr_t)bptr;
    ASYNC_B128(boff[0], gb, 0);  ASYNC_B128(boff[0], gb, 16);
  }
  int nk = K / GBK;
  for (int kt = 0; kt < nk; ++kt) {
    WAIT_ASYNC();        // our async loads for buffer `cur` complete
    __syncthreads();     // everyone's complete -> tile ready
    int cur = kt & 1;
    if (kt + 1 < nk) {   // prefetch next tile into other buffer (overlaps WMMA below)
      int k1 = (kt + 1) * GBK;
      unsigned long long ga = (unsigned long long)(uintptr_t)(aptr + k1);
      ASYNC_B128(aoff[cur ^ 1], ga, 0);  ASYNC_B128(aoff[cur ^ 1], ga, 16);
      ASYNC_B128(aoff[cur ^ 1], ga, 32); ASYNC_B128(aoff[cur ^ 1], ga, 48);
      unsigned long long gb = (unsigned long long)(uintptr_t)(bptr + k1);
      ASYNC_B128(boff[cur ^ 1], gb, 0);  ASYNC_B128(boff[cur ^ 1], gb, 16);
    }
    const unsigned short* ac = as[cur];
    const unsigned short* bc = bs[cur];
    FragB a[2], bfr[4];
    for (int mt = 0; mt < 2; ++mt) {
      int m = wave * 32 + mt * 16 + l16;
      for (int p = 0; p < 8; ++p) a[mt].u[p] = *(const unsigned*)(ac + m * ALD + kpA(p, half));
    }
    for (int nt = 0; nt < 4; ++nt) {
      int n = nt * 16 + l16;
      for (int p = 0; p < 8; ++p) bfr[nt].u[p] = *(const unsigned*)(bc + n * BLD + kpB(p, half));
    }
    for (int mt = 0; mt < 2; ++mt)
      for (int nt = 0; nt < 4; ++nt)
        acc[mt][nt].v = __builtin_amdgcn_wmma_f32_16x16x32_bf16(
            false, a[mt].v, false, bfr[nt].v, (short)0, acc[mt][nt].v, false, false);
    // no end barrier needed: next iteration's top barrier orders buffer reuse
  }
  for (int mt = 0; mt < 2; ++mt) {
    for (int nt = 0; nt < 4; ++nt) {
      int n = bn0 + nt * 16 + l16;
      float bv = bias ? bias[n] : 0.f;
      for (int i = 0; i < 8; ++i) {
        int m = bm0 + wave * 32 + mt * 16 + i + half * 8;
        float v = acc[mt][nt].f[i] * scale + bv;
        if (relu) v = fmaxf(v, 0.f);
        size_t off = (size_t)m * N + n;
        if (outF) outF[off] = v;
        if (outB) outB[off] = f2bf(v);
      }
    }
  }
}

// ---------------- qrel = Q @ rel_emb^T : [B*H*S, NRELP] bf16 ----------------
__global__ __launch_bounds__(128) void k_qrel(const unsigned short* __restrict__ qb,
                                              const unsigned short* __restrict__ relb, // [257,64] bf16
                                              unsigned short* __restrict__ qrel) {     // [B*H*S, 272] bf16
  int tid = threadIdx.x;
  int wave = tid >> 5, lane = tid & 31;
  int half = lane >> 4, l16 = lane & 15;
  int q0 = blockIdx.x * 64;
  int bh = blockIdx.y;
  int b = bh / H_, h = bh % H_;
  FragB aq0, aq1;
  {
    int m = q0 + wave * 16 + l16;
    const unsigned short* qrow = qb + ((size_t)(b * S_ + m)) * D_ + h * HD_;
    for (int p = 0; p < 8; ++p) {
      aq0.u[p] = *(const unsigned*)(qrow + kpA(p, half));
      aq1.u[p] = *(const unsigned*)(qrow + 32 + kpA(p, half));
    }
  }
  for (int j0 = 0; j0 < NRELP_; j0 += 16) {
    int j = j0 + l16; if (j > NREL_ - 1) j = NREL_ - 1;
    FragB b0, b1;
    const unsigned short* rrow = relb + (size_t)j * HD_;
    for (int p = 0; p < 8; ++p) {
      b0.u[p] = *(const unsigned*)(rrow + kpB(p, half));
      b1.u[p] = *(const unsigned*)(rrow + 32 + kpB(p, half));
    }
    FragF c; for (int e = 0; e < 8; ++e) c.f[e] = 0.f;
    c.v = __builtin_amdgcn_wmma_f32_16x16x32_bf16(false, aq0.v, false, b0.v, (short)0, c.v, false, false);
    c.v = __builtin_amdgcn_wmma_f32_16x16x32_bf16(false, aq1.v, false, b1.v, (short)0, c.v, false, false);
    size_t base = (size_t)bh * S_ + q0 + wave * 16;
    for (int i = 0; i < 8; ++i) {
      int r = i + half * 8;
      qrel[(base + r) * NRELP_ + j0 + l16] = f2bf(c.f[i]);
    }
  }
}

// ---------------- flash attention with relative bias ----------------
#define QRLD 260
#define NKT  (S_ / 32)
__global__ __launch_bounds__(128) void k_attn(const unsigned short* __restrict__ qb,
                                              const unsigned short* __restrict__ kb, // pre-scaled by 1/sqrt(HD)
                                              const unsigned short* __restrict__ vb,
                                              const unsigned short* __restrict__ qrel,
                                              unsigned short* __restrict__ ctx) {
  __shared__ __align__(16) unsigned short qr_s[64 * QRLD];   // 33.3 KB
  __shared__ __align__(16) unsigned short ksb[2][32 * 72];   // K tile [key][d], double buffered
  __shared__ __align__(16) unsigned short vtb[2][HD_ * 40];  // V tile transposed [d][key], double buffered
  __shared__ __align__(16) unsigned short pr[4 * 16 * 36];   // per-wave probs staging
  int tid = threadIdx.x;
  int wave = tid >> 5, lane = tid & 31;
  int half = lane >> 4, l16 = lane & 15;
  int q0 = blockIdx.x * 64;
  int bh = blockIdx.y;
  int b = bh / H_, h = bh % H_;

  int key = tid >> 2, ch = (tid & 3) * 16;
  unsigned koff[2] = { ldsoff(&ksb[0][key * 72 + ch]), ldsoff(&ksb[1][key * 72 + ch]) };

  // stage K tile via async global->LDS, V tile via VGPR transpose
  auto stage_kv = [&](int buf, int k0) {
    const unsigned short* kg = kb + ((size_t)(b * S_ + k0 + key)) * D_ + h * HD_ + ch;
    unsigned long long ga = (unsigned long long)(uintptr_t)kg;
    ASYNC_B128(koff[buf], ga, 0); ASYNC_B128(koff[buf], ga, 16);
    U4x2 vsrc;
    const uint4* vs = (const uint4*)(vb + ((size_t)(b * S_ + k0 + key)) * D_ + h * HD_ + ch);
    vsrc.q[0] = vs[0]; vsrc.q[1] = vs[1];
    unsigned short* vt = vtb[buf];
    for (int i = 0; i < 16; ++i) vt[(ch + i) * 40 + key] = vsrc.s[i];
  };

  // preload qrel rows for our 64 queries
  {
    size_t qrbase = (size_t)bh * S_ + q0;
    int r = tid & 63, c0 = tid >> 6;
    for (int c = c0; c < NREL_; c += 2)
      qr_s[r * QRLD + c] = qrel[(qrbase + r) * NRELP_ + c];
  }
  // Q A-fragments (held in registers whole kernel)
  FragB aq0, aq1;
  {
    int m = q0 + wave * 16 + l16;
    const unsigned short* qrow = qb + ((size_t)(b * S_ + m)) * D_ + h * HD_;
    for (int p = 0; p < 8; ++p) {
      aq0.u[p] = *(const unsigned*)(qrow + kpA(p, half));
      aq1.u[p] = *(const unsigned*)(qrow + 32 + kpA(p, half));
    }
  }
  FragF o[4];
  for (int f = 0; f < 4; ++f) for (int e = 0; e < 8; ++e) o[f].f[e] = 0.f;
  float m_run[8], l_run[8];
  for (int i = 0; i < 8; ++i) { m_run[i] = -3.0e38f; l_run[i] = 0.f; }

  stage_kv(0, 0);   // prologue

  unsigned short* prw = pr + wave * 16 * 36;
  for (int kt = 0; kt < NKT; ++kt) {
    int k0 = kt * 32;
    WAIT_ASYNC();
    __syncthreads();                     // tile(cur) + qrel ready everywhere
    int cur = kt & 1;
    if (kt + 1 < NKT) stage_kv(cur ^ 1, k0 + 32);   // overlap with compute below
    const unsigned short* ks = ksb[cur];
    const unsigned short* vt = vtb[cur];
    // scores = Q @ K^T (two 16x16 tiles covering 32 keys)
    FragF s0, s1;
    for (int e = 0; e < 8; ++e) { s0.f[e] = 0.f; s1.f[e] = 0.f; }
    {
      FragB bk;
      int n = l16;
      for (int p = 0; p < 8; ++p) bk.u[p] = *(const unsigned*)(ks + n * 72 + kpB(p, half));
      s0.v = __builtin_amdgcn_wmma_f32_16x16x32_bf16(false, aq0.v, false, bk.v, (short)0, s0.v, false, false);
      for (int p = 0; p < 8; ++p) bk.u[p] = *(const unsigned*)(ks + n * 72 + 32 + kpB(p, half));
      s0.v = __builtin_amdgcn_wmma_f32_16x16x32_bf16(false, aq1.v, false, bk.v, (short)0, s0.v, false, false);
      n = 16 + l16;
      for (int p = 0; p < 8; ++p) bk.u[p] = *(const unsigned*)(ks + n * 72 + kpB(p, half));
      s1.v = __builtin_amdgcn_wmma_f32_16x16x32_bf16(false, aq0.v, false, bk.v, (short)0, s1.v, false, false);
      for (int p = 0; p < 8; ++p) bk.u[p] = *(const unsigned*)(ks + n * 72 + 32 + kpB(p, half));
      s1.v = __builtin_amdgcn_wmma_f32_16x16x32_bf16(false, aq1.v, false, bk.v, (short)0, s1.v, false, false);
    }
    // rel bias + online softmax
    for (int i = 0; i < 8; ++i) {
      int r = i + half * 8;
      int rl = wave * 16 + r;
      int qg = q0 + rl;
      int n0 = k0 + l16, n1 = n0 + 16;
      int j0 = n0 - qg; j0 = j0 < -MAXREL_ ? -MAXREL_ : (j0 > MAXREL_ ? MAXREL_ : j0); j0 += MAXREL_;
      int j1 = n1 - qg; j1 = j1 < -MAXREL_ ? -MAXREL_ : (j1 > MAXREL_ ? MAXREL_ : j1); j1 += MAXREL_;
      float v0 = s0.f[i] + bf2f(qr_s[rl * QRLD + j0]);
      float v1 = s1.f[i] + bf2f(qr_s[rl * QRLD + j1]);
      float mx = fmaxf(v0, v1);
      for (int d = 1; d < 16; d <<= 1) mx = fmaxf(mx, __shfl_xor(mx, d, 16));
      float mnew = fmaxf(m_run[i], mx);
      float corr = __expf(m_run[i] - mnew);
      float p0 = __expf(v0 - mnew), p1 = __expf(v1 - mnew);
      float sum = p0 + p1;
      for (int d = 1; d < 16; d <<= 1) sum += __shfl_xor(sum, d, 16);
      l_run[i] = l_run[i] * corr + sum;
      m_run[i] = mnew;
      for (int f = 0; f < 4; ++f) o[f].f[i] *= corr;
      prw[r * 36 + l16] = f2bf(p0);
      prw[r * 36 + 16 + l16] = f2bf(p1);
    }
    // wave-internal LDS store->load ordering for probs re-layout
    WAIT_DS();
    FragB ap;
    for (int p = 0; p < 8; ++p) ap.u[p] = *(const unsigned*)(prw + l16 * 36 + kpA(p, half));
    FragB bv;
    for (int f = 0; f < 4; ++f) {
      int n = f * 16 + l16;
      for (int p = 0; p < 8; ++p) bv.u[p] = *(const unsigned*)(vt + n * 40 + kpB(p, half));
      o[f].v = __builtin_amdgcn_wmma_f32_16x16x32_bf16(false, ap.v, false, bv.v, (short)0, o[f].v, false, false);
    }
    // no end barrier: next iteration's top barrier orders double-buffer reuse
  }
  // normalize and store ctx back to [B,S,D] (col = h*64 + d)
  for (int i = 0; i < 8; ++i) {
    int r = i + half * 8;
    float inv = 1.0f / l_run[i];
    int sg = q0 + wave * 16 + r;
    unsigned short* dst = ctx + ((size_t)(b * S_ + sg)) * D_ + h * HD_;
    for (int f = 0; f < 4; ++f) dst[f * 16 + l16] = f2bf(o[f].f[i] * inv);
  }
}

// ---------------- residual + layernorm ----------------
__global__ __launch_bounds__(256) void k_ln(const float* __restrict__ x, const float* __restrict__ delta,
                                            const float* __restrict__ g, const float* __restrict__ bta,
                                            float* __restrict__ outF, unsigned short* __restrict__ outB) {
  __shared__ float red[16];
  int row = blockIdx.x, tid = threadIdx.x;
  float v0 = x[(size_t)row * D_ + tid]       + (delta ? delta[(size_t)row * D_ + tid]       : 0.f);
  float v1 = x[(size_t)row * D_ + tid + 256] + (delta ? delta[(size_t)row * D_ + tid + 256] : 0.f);
  float s = v0 + v1, s2 = v0 * v0 + v1 * v1;
  for (int d = 1; d < 32; d <<= 1) { s += __shfl_xor(s, d, 32); s2 += __shfl_xor(s2, d, 32); }
  int wave = tid >> 5;
  if ((tid & 31) == 0) { red[wave] = s; red[8 + wave] = s2; }
  __syncthreads();
  float ts = 0.f, ts2 = 0.f;
  for (int i = 0; i < 8; ++i) { ts += red[i]; ts2 += red[8 + i]; }
  float mean = ts / (float)D_;
  float var = ts2 / (float)D_ - mean * mean;
  float inv = rsqrtf(var + EPS_);
  for (int j = 0; j < 2; ++j) {
    int c = tid + 256 * j;
    float v = j ? v1 : v0;
    float y = (v - mean) * inv * g[c] + bta[c];
    if (outF) outF[(size_t)row * D_ + c] = y;
    if (outB) outB[(size_t)row * D_ + c] = f2bf(y);
  }
}

// ---------------- head second layer: out = p1o @ p2_w + p2_b ----------------
__global__ void k_head(const unsigned short* __restrict__ p1o, const float* __restrict__ w2,
                       const float* __restrict__ b2, float* __restrict__ out) {
  int row = blockIdx.x * 256 + threadIdx.x;
  float acc = b2[0];
  const unsigned short* src = p1o + (size_t)row * 256;
  for (int i = 0; i < 256; ++i) acc += bf2f(src[i]) * w2[i];
  out[row] = acc;
}

// ---------------- host orchestration ----------------
extern "C" void kernel_launch(void* const* d_in, const int* in_sizes, int n_in,
                              void* d_out, int out_size, void* d_ws, size_t ws_size,
                              hipStream_t stream) {
  (void)in_sizes; (void)n_in; (void)out_size; (void)ws_size;
  const float* x    = (const float*)d_in[0];
  const float* in_w = (const float*)d_in[1];
  const float* in_b = (const float*)d_in[2];
  const float* Wq   = (const float*)d_in[3];
  const float* Wk   = (const float*)d_in[4];
  const float* Wv   = (const float*)d_in[5];
  const float* Wo   = (const float*)d_in[6];
  const float* bo   = (const float*)d_in[7];
  const float* rel  = (const float*)d_in[8];
  const float* w1   = (const float*)d_in[9];
  const float* b1   = (const float*)d_in[10];
  const float* w2   = (const float*)d_in[11];
  const float* b2   = (const float*)d_in[12];
  const float* ln1g = (const float*)d_in[13];
  const float* ln1b = (const float*)d_in[14];
  const float* ln2g = (const float*)d_in[15];
  const float* ln2b = (const float*)d_in[16];
  const float* ong  = (const float*)d_in[17];
  const float* onb  = (const float*)d_in[18];
  const float* p1w  = (const float*)d_in[19];
  const float* p1b  = (const float*)d_in[20];
  const float* p2w  = (const float*)d_in[21];
  const float* p2b  = (const float*)d_in[22];

  char* ws = (char*)d_ws;
  size_t off = 0;
  auto alloc = [&](size_t bytes) -> char* {
    char* p = ws + off;
    off += (bytes + 255) & ~(size_t)255;
    return p;
  };
  unsigned short* WqT  = (unsigned short*)alloc((size_t)L_ * D_ * D_ * 2);
  unsigned short* WkT  = (unsigned short*)alloc((size_t)L_ * D_ * D_ * 2);
  unsigned short* WvT  = (unsigned short*)alloc((size_t)L_ * D_ * D_ * 2);
  unsigned short* WoT  = (unsigned short*)alloc((size_t)L_ * D_ * D_ * 2);
  unsigned short* w1T  = (unsigned short*)alloc((size_t)L_ * D_ * FF_ * 2);
  unsigned short* w2T  = (unsigned short*)alloc((size_t)L_ * FF_ * D_ * 2);
  unsigned short* p1T  = (unsigned short*)alloc((size_t)D_ * (D_/2) * 2);
  unsigned short* relb = (unsigned short*)alloc((size_t)L_ * NREL_ * HD_ * 2);
  float*          hbuf = (float*)alloc((size_t)ROWS_ * D_ * 4);
  unsigned short* hb   = (unsigned short*)alloc((size_t)ROWS_ * D_ * 2);
  unsigned short* qbuf = (unsigned short*)alloc((size_t)ROWS_ * D_ * 2);
  unsigned short* kbuf = (unsigned short*)alloc((size_t)ROWS_ * D_ * 2);
  unsigned short* vbuf = (unsigned short*)alloc((size_t)ROWS_ * D_ * 2);
  unsigned short* ctx  = (unsigned short*)alloc((size_t)ROWS_ * D_ * 2);
  float*          tmpF = (float*)alloc((size_t)ROWS_ * D_ * 4);
  unsigned short* ffm  = (unsigned short*)alloc((size_t)ROWS_ * FF_ * 2);
  unsigned short* qrel = (unsigned short*)alloc((size_t)B_ * H_ * S_ * NRELP_ * 2);
  unsigned short* p1o  = (unsigned short*)alloc((size_t)ROWS_ * (D_/2) * 2);

  // --- weight conversion (bf16, B stored transposed [N,K]) ---
  int nDD = D_ * D_;
  for (int l = 0; l < L_; ++l) {
    hipLaunchKernelGGL(k_transpose_bf16, dim3((nDD + 255) / 256), dim3(256), 0, stream,
                       Wq + (size_t)l * nDD, WqT + (size_t)l * nDD, D_, D_);
    hipLaunchKernelGGL(k_transpose_bf16, dim3((nDD + 255) / 256), dim3(256), 0, stream,
                       Wk + (size_t)l * nDD, WkT + (size_t)l * nDD, D_, D_);
    hipLaunchKernelGGL(k_transpose_bf16, dim3((nDD + 255) / 256), dim3(256), 0, stream,
                       Wv + (size_t)l * nDD, WvT + (size_t)l * nDD, D_, D_);
    hipLaunchKernelGGL(k_transpose_bf16, dim3((nDD + 255) / 256), dim3(256), 0, stream,
                       Wo + (size_t)l * nDD, WoT + (size_t)l * nDD, D_, D_);
    int nDF = D_ * FF_;
    hipLaunchKernelGGL(k_transpose_bf16, dim3((nDF + 255) / 256), dim3(256), 0, stream,
                       w1 + (size_t)l * nDF, w1T + (size_t)l * nDF, D_, FF_);
    hipLaunchKernelGGL(k_transpose_bf16, dim3((nDF + 255) / 256), dim3(256), 0, stream,
                       w2 + (size_t)l * nDF, w2T + (size_t)l * nDF, FF_, D_);
  }
  hipLaunchKernelGGL(k_transpose_bf16, dim3((D_ * (D_/2) + 255) / 256), dim3(256), 0, stream,
                     p1w, p1T, D_, D_ / 2);
  hipLaunchKernelGGL(k_copy_bf16, dim3((L_ * NREL_ * HD_ + 255) / 256), dim3(256), 0, stream,
                     rel, relb, L_ * NREL_ * HD_);

  // --- input projection + PE ---
  hipLaunchKernelGGL(k_input, dim3(ROWS_), dim3(128), 0, stream, x, in_w, in_b, hbuf, hb);

  dim3 gD(D_ / GBN, ROWS_ / GBM);    // N=512
  dim3 gFF(FF_ / GBN, ROWS_ / GBM);  // N=2048
  dim3 gP1((D_/2) / GBN, ROWS_ / GBM);
  dim3 gAtt(S_ / 64, B_ * H_);

  for (int l = 0; l < L_; ++l) {
    const unsigned short* wq = WqT + (size_t)l * nDD;
    const unsigned short* wk = WkT + (size_t)l * nDD;
    const unsigned short* wv = WvT + (size_t)l * nDD;
    const unsigned short* wo = WoT + (size_t)l * nDD;
    // Q, K (pre-scaled by 1/sqrt(HD)), V projections (no bias)
    hipLaunchKernelGGL(k_gemm, gD, dim3(128), 0, stream, hb, wq, (const float*)nullptr, 1.0f, 0,
                       ROWS_, D_, D_, (float*)nullptr, qbuf);
    hipLaunchKernelGGL(k_gemm, gD, dim3(128), 0, stream, hb, wk, (const float*)nullptr, 0.125f, 0,
                       ROWS_, D_, D_, (float*)nullptr, kbuf);
    hipLaunchKernelGGL(k_gemm, gD, dim3(128), 0, stream, hb, wv, (const float*)nullptr, 1.0f, 0,
                       ROWS_, D_, D_, (float*)nullptr, vbuf);
    // qrel = Q @ rel_emb^T
    hipLaunchKernelGGL(k_qrel, gAtt, dim3(128), 0, stream, qbuf,
                       relb + (size_t)l * NREL_ * HD_, qrel);
    // flash attention
    hipLaunchKernelGGL(k_attn, gAtt, dim3(128), 0, stream, qbuf, kbuf, vbuf, qrel, ctx);
    // output projection + bias
    hipLaunchKernelGGL(k_gemm, gD, dim3(128), 0, stream, ctx, wo, bo + (size_t)l * D_, 1.0f, 0,
                       ROWS_, D_, D_, tmpF, (unsigned short*)nullptr);
    // h = LN(h + attn_out)
    hipLaunchKernelGGL(k_ln, dim3(ROWS_), dim3(256), 0, stream, hbuf, tmpF,
                       ln1g + (size_t)l * D_, ln1b + (size_t)l * D_, hbuf, hb);
    // FFN
    hipLaunchKernelGGL(k_gemm, gFF, dim3(128), 0, stream, hb, w1T + (size_t)l * D_ * FF_,
                       b1 + (size_t)l * FF_, 1.0f, 1, ROWS_, FF_, D_, (float*)nullptr, ffm);
    hipLaunchKernelGGL(k_gemm, gD, dim3(128), 0, stream, ffm, w2T + (size_t)l * D_ * FF_,
                       b2 + (size_t)l * D_, 1.0f, 0, ROWS_, D_, FF_, tmpF, (unsigned short*)nullptr);
    // h = LN(h + ff)
    hipLaunchKernelGGL(k_ln, dim3(ROWS_), dim3(256), 0, stream, hbuf, tmpF,
                       ln2g + (size_t)l * D_, ln2b + (size_t)l * D_, hbuf, hb);
  }
  // final LN
  hipLaunchKernelGGL(k_ln, dim3(ROWS_), dim3(256), 0, stream, hbuf, (const float*)nullptr,
                     ong, onb, (float*)nullptr, hb);
  // head
  hipLaunchKernelGGL(k_gemm, gP1, dim3(128), 0, stream, hb, p1T, p1b, 1.0f, 1,
                     ROWS_, D_ / 2, D_, (float*)nullptr, p1o);
  hipLaunchKernelGGL(k_head, dim3(ROWS_ / 256), dim3(256), 0, stream, p1o, p2w, p2b, (float*)d_out);
}
